// AttentionDecoder_4956392260118
// MI455X (gfx1250) — matrix operand order repaired and checked by reading, not compiled, not run on hardware
//
#include <hip/hip_runtime.h>
#include <cstdint>
#include <cstddef>

// ---------------------------------------------------------------------------
// AttentionDecoder for gfx1250 (MI455X).
//  - V_WMMA_F32_16X16X32_F16, wave32, 4 N-tiles per wave (register reuse of A)
//  - A k-slab staged to LDS via GLOBAL_LOAD_ASYNC_TO_LDS_B128 (double buffer,
//    ASYNCcnt), shared by all 4 waves of the block
//  - B weight stream forced to addrspace(1) so it lowers to global_load_b128
//    (LOADcnt only; flat ops would couple LOADcnt with DScnt and the LDS path)
// ---------------------------------------------------------------------------

#define Bv   32
#define Tv   128
#define Sv   128
#define Dv   512
#define Hv   512
#define Vv   16000
#define G4H  2048   // 4*H
#define H2   1024   // 2*H

typedef __attribute__((ext_vector_type(16))) _Float16 v16h;
typedef __attribute__((ext_vector_type(8)))  float    v8f;
typedef __attribute__((__vector_size__(16))) int      v4i_;

// addrspace(1) views only exist in the device pass (host chokes on AS-qualified
// class references; plain vector/scalar types assign fine via builtin ops)
#if defined(__HIP_DEVICE_COMPILE__)
#define AS1 __attribute__((address_space(1)))
#else
#define AS1
#endif
typedef AS1 const v4i_     g_cv4i;
typedef AS1 const float    g_cfloat;
typedef AS1       float    g_float;
typedef AS1 const _Float16 g_chalf;

union FragU { v16h v; v4i_ q[2]; };

#if defined(__HIP_DEVICE_COMPILE__) && __has_builtin(__builtin_amdgcn_global_load_async_to_lds_b128)
#define HAVE_ASYNC_LDS 1
#else
#define HAVE_ASYNC_LDS 0
#endif

#if defined(__HIP_DEVICE_COMPILE__) && __has_builtin(__builtin_amdgcn_s_wait_asynccnt)
#define WAIT_ASYNC() __builtin_amdgcn_s_wait_asynccnt(0)
#elif defined(__HIP_DEVICE_COMPILE__)
#define WAIT_ASYNC() asm volatile("s_wait_asynccnt 0x0" ::: "memory")
#else
#define WAIT_ASYNC()
#endif

static __device__ __forceinline__ float sigmoidf_(float x) {
    return 1.0f / (1.0f + __expf(-x));
}

// -------------------------- conversion kernels -----------------------------

__global__ void k_convert(const float* __restrict__ src, _Float16* __restrict__ dst, int n) {
    int i = blockIdx.x * blockDim.x + threadIdx.x;
    if (i < n) dst[i] = (_Float16)src[i];
}

// dst[k*Hv + h] = src[h*Hv + k]
__global__ void k_convT(const float* __restrict__ src, _Float16* __restrict__ dst) {
    int i = blockIdx.x * blockDim.x + threadIdx.x;
    if (i < Hv * Hv) {
        int k = i / Hv, h = i % Hv;
        dst[i] = (_Float16)src[h * Hv + k];
    }
}

__global__ void k_bias2(const float* __restrict__ a, const float* __restrict__ b,
                        float* __restrict__ o, int n) {
    int i = blockIdx.x * blockDim.x + threadIdx.x;
    if (i < n) o[i] = a[i] + b[i];
}

// emb_h[t*B*D + b*D + d] = f16(embed[tok(t,b)*D + d])
__global__ void k_gather(const int* __restrict__ x, const int* __restrict__ teacher,
                         const float* __restrict__ embed, _Float16* __restrict__ emb_h) {
    int i = blockIdx.x * blockDim.x + threadIdx.x;
    if (i >= Tv * Bv * Dv) return;
    int t = i / (Bv * Dv);
    int r = i % (Bv * Dv);
    int b = r / Dv;
    int d = r % Dv;
    int tok = (t == 0) ? x[b * Tv] : teacher[b * Tv + (t - 1)];
    emb_h[i] = (_Float16)embed[(size_t)tok * Dv + d];
}

__global__ void k_init(float* __restrict__ cx, _Float16* __restrict__ hx_h) {
    int i = blockIdx.x * blockDim.x + threadIdx.x;
    if (i < Bv * Hv) { cx[i] = 0.0f; hx_h[i] = (_Float16)0.0f; }
}

// ------------------------------ WMMA GEMM ----------------------------------
// out[m,n] = sum_k A[m,k]*W[n,k] (+bias[n]) (+Cadd[m,n])
// A: f16 [M,K] (lda), W: f16 [N,K] (ldw). M mult of 16, N mult of 64, K mult
// of 32. Block = 128 threads (4 waves), one M-tile per block, 4 N-tiles per
// wave (16 per block). A k-slab (16x32 f16 = 1KB) double-buffered in LDS via
// async copy issued by wave 0; B fragments stream from L2 as global_load_b128.

__global__ void __launch_bounds__(128)
k_gemm_wmma(const _Float16* __restrict__ A, int lda,
            const _Float16* __restrict__ W, int ldw,
            const float* __restrict__ bias,
            const float* __restrict__ Cadd, int ldca,
            float* __restrict__ out, long long ldc,
            int M, int N, int K) {
    __shared__ alignas(16) _Float16 ldsA[2][16 * 32];

    int groups = N >> 6;            // wave groups of 4 N-tiles
    int gb     = (groups + 3) >> 2; // groups per block (4 waves)
    int mt     = blockIdx.x / gb;
    int gbi    = blockIdx.x % gb;

    int wid  = threadIdx.x >> 5;
    int lane = threadIdx.x & 31;
    int lrow = lane & 15;
    int half = lane >> 4;

    int grp = gbi * 4 + wid;
    if (grp >= groups) grp = groups - 1;   // duplicate tail work (same values)
    int nt0 = grp * 4;

    // A slab async stage: 1KB = 2 x (32 lanes x 16B)
    auto issueA = [&](int kb, int buf) {
        if (wid == 0) {
#pragma unroll
            for (int c = 0; c < 2; ++c) {
                int o    = c * 512 + lane * 16;      // byte offset in slab
                int row  = o >> 6;                   // 64B per row
                int colb = o & 63;
                const _Float16* gp = A + (size_t)(mt * 16 + row) * lda + kb + (colb >> 1);
                char* lp = (char*)(&ldsA[buf][0]) + o;
#if HAVE_ASYNC_LDS
                __builtin_amdgcn_global_load_async_to_lds_b128(
                    (__attribute__((address_space(1))) v4i_*)gp,
                    (__attribute__((address_space(3))) v4i_*)lp, 0, 0);
#else
                *(v4i_*)lp = *(const v4i_*)gp;
#endif
            }
        }
    };

    // B pointers in global addrspace: lane = column (nt*16+lrow), contiguous
    // 16 K elements per half-wave
    g_chalf* wp[4];
#pragma unroll
    for (int j = 0; j < 4; ++j)
        wp[j] = (g_chalf*)(W + (size_t)((nt0 + j) * 16 + lrow) * ldw + half * 16);

    v8f acc[4];
#pragma unroll
    for (int j = 0; j < 4; ++j) acc[j] = v8f{0.f, 0.f, 0.f, 0.f, 0.f, 0.f, 0.f, 0.f};

    issueA(0, 0);
    int cur = 0;
    for (int kb = 0; kb < K; kb += 32) {
#if HAVE_ASYNC_LDS
        if (wid == 0) WAIT_ASYNC();
#endif
        __syncthreads();                 // slab[cur] ready; slab[cur^1] free
        if (kb + 32 < K) issueA(kb + 32, cur ^ 1);

        FragU a;
        const char* ab = (const char*)(&ldsA[cur][0]) + lrow * 64 + half * 32;
        a.q[0] = *(const v4i_*)(ab);         // K kb+half*16 .. +8
        a.q[1] = *(const v4i_*)(ab + 16);    // K kb+half*16+8 .. +8

#pragma unroll
        for (int j = 0; j < 4; ++j) {
            FragU b;
            b.q[0] = *(g_cv4i*)(wp[j]);
            b.q[1] = *(g_cv4i*)(wp[j] + 8);
            wp[j] += 32;
            __builtin_prefetch((const void*)(wp[j] + 96), 0, 0);
            acc[j] = __builtin_amdgcn_wmma_f32_16x16x32_f16(
                         false, a.v, false, b.v, (short)0, acc[j], false, false);
        }
        cur ^= 1;
    }

#pragma unroll
    for (int j = 0; j < 4; ++j) {
        int n = (nt0 + j) * 16 + lrow;
        float badd = bias ? *(g_cfloat*)(bias + n) : 0.0f;
#pragma unroll
        for (int v = 0; v < 8; ++v) {
            int m = mt * 16 + v + 8 * half;   // C/D layout: VGPR v -> M = v + 8*half
            float val = acc[j][v] + badd;
            if (Cadd) val += *(g_cfloat*)(Cadd + (size_t)m * ldca + n);
            *(g_float*)(out + (size_t)m * ldc + n) = val;
        }
    }
}

// ------------------------------ LSTM cell ----------------------------------

__global__ void k_cell(const float* __restrict__ gates, float* __restrict__ cx,
                       _Float16* __restrict__ hx_h, _Float16* __restrict__ hcat_h) {
    int i = blockIdx.x * blockDim.x + threadIdx.x;   // i = b*H + h
    if (i >= Bv * Hv) return;
    int b = i >> 9;
    int h = i & 511;
    const float* g = gates + (size_t)b * G4H;
    float ig = sigmoidf_(g[h]);
    float fg = sigmoidf_(g[512 + h]);
    float gg = tanhf(g[1024 + h]);
    float og = sigmoidf_(g[1536 + h]);
    float c  = fg * cx[i] + ig * gg;
    cx[i] = c;
    float hv = og * tanhf(c);
    _Float16 hh = (_Float16)hv;
    hx_h[i] = hh;
    hcat_h[b * H2 + Hv + h] = hh;     // second half of [cxt, hx]
}

// ------------------------- additive attention ------------------------------
// one block per batch row b, 128 threads (== S)

__global__ void k_attn(const float* __restrict__ q, const float* __restrict__ keys,
                       const float* __restrict__ z, const float* __restrict__ v_att,
                       _Float16* __restrict__ hcat_h) {
    __shared__ float sq[Hv];
    __shared__ float sv[Hv];
    __shared__ float sw[Sv];
    __shared__ float sred[Sv];

    int b   = blockIdx.x;
    int tid = threadIdx.x;

    for (int h = tid; h < Hv; h += blockDim.x) {
        sq[h] = q[b * Hv + h];
        sv[h] = v_att[h];
    }
    __syncthreads();

    const float* krow = keys + ((size_t)b * Sv + tid) * Hv;
    float sc = 0.0f;
    for (int h = 0; h < Hv; ++h) sc += tanhf(krow[h] + sq[h]) * sv[h];

    sred[tid] = sc;
    __syncthreads();
    for (int off = Sv >> 1; off > 0; off >>= 1) {
        if (tid < off) sred[tid] = fmaxf(sred[tid], sred[tid + off]);
        __syncthreads();
    }
    float mx = sred[0];
    __syncthreads();
    float e = __expf(sc - mx);
    sred[tid] = e;
    __syncthreads();
    for (int off = Sv >> 1; off > 0; off >>= 1) {
        if (tid < off) sred[tid] += sred[tid + off];
        __syncthreads();
    }
    float denom = sred[0];
    __syncthreads();
    sw[tid] = e / denom;
    __syncthreads();

    for (int h = tid; h < Hv; h += blockDim.x) {
        float accv = 0.0f;
        const float* zb = z + (size_t)b * Sv * Hv + h;
        for (int s = 0; s < Sv; ++s) accv += sw[s] * zb[(size_t)s * Hv];
        hcat_h[b * H2 + h] = (_Float16)accv;   // first half of [cxt, hx]
    }
}

// ------------------------------ launcher -----------------------------------

extern "C" void kernel_launch(void* const* d_in, const int* in_sizes, int n_in,
                              void* d_out, int out_size, void* d_ws, size_t ws_size,
                              hipStream_t stream) {
    (void)in_sizes; (void)n_in; (void)out_size; (void)ws_size;

    const int*   x      = (const int*)  d_in[0];
    const int*   teach  = (const int*)  d_in[1];
    const float* z      = (const float*)d_in[2];
    const float* embed  = (const float*)d_in[3];
    const float* W_ih   = (const float*)d_in[4];
    const float* b_ih   = (const float*)d_in[5];
    const float* W_hh   = (const float*)d_in[6];
    const float* b_hh   = (const float*)d_in[7];
    const float* Wq     = (const float*)d_in[8];
    const float* Wk     = (const float*)d_in[9];
    const float* v_att  = (const float*)d_in[10];
    const float* W_proj = (const float*)d_in[11];
    const float* b_proj = (const float*)d_in[12];
    float* out = (float*)d_out;

    char* p = (char*)d_ws;
    auto carve = [&](size_t bytes) -> char* {
        char* r = p;
        p += (bytes + 255) & ~(size_t)255;
        return r;
    };
    _Float16* Wih_h  = (_Float16*)carve((size_t)G4H * Dv * 2);
    _Float16* Whh_h  = (_Float16*)carve((size_t)G4H * Hv * 2);
    _Float16* Wq_h   = (_Float16*)carve((size_t)Hv * Hv * 2);
    _Float16* WkT_h  = (_Float16*)carve((size_t)Hv * Hv * 2);
    _Float16* Wp_h   = (_Float16*)carve((size_t)Vv * H2 * 2);
    _Float16* z_h    = (_Float16*)carve((size_t)Bv * Sv * Hv * 2);
    _Float16* emb_h  = (_Float16*)carve((size_t)Tv * Bv * Dv * 2);
    float*    bsum   = (float*)   carve((size_t)G4H * 4);
    float*    Gpre   = (float*)   carve((size_t)Tv * Bv * G4H * 4);
    float*    keys   = (float*)   carve((size_t)Bv * Sv * Hv * 4);
    float*    gates  = (float*)   carve((size_t)Bv * G4H * 4);
    float*    qbuf   = (float*)   carve((size_t)Bv * Hv * 4);
    float*    cx     = (float*)   carve((size_t)Bv * Hv * 4);
    _Float16* hx_h   = (_Float16*)carve((size_t)Bv * Hv * 2);
    _Float16* hcat_h = (_Float16*)carve((size_t)Bv * H2 * 2);

    auto conv = [&](const float* s, _Float16* d, int n) {
        k_convert<<<(n + 255) / 256, 256, 0, stream>>>(s, d, n);
    };
    auto gemm = [&](const _Float16* A, int lda, const _Float16* W, int ldw,
                    const float* bias, const float* Cadd, int ldca,
                    float* o, long long ldc, int M, int N, int K) {
        int groups = N / 64;                 // N must be multiple of 64
        int gb     = (groups + 3) / 4;
        int blocks = (M / 16) * gb;
        k_gemm_wmma<<<blocks, 128, 0, stream>>>(A, lda, W, ldw, bias, Cadd, ldca,
                                                o, ldc, M, N, K);
    };

    // one-time precompute
    conv(W_ih,   Wih_h, G4H * Dv);
    conv(W_hh,   Whh_h, G4H * Hv);
    conv(Wq,     Wq_h,  Hv * Hv);
    conv(W_proj, Wp_h,  Vv * H2);
    conv(z,      z_h,   Bv * Sv * Hv);
    k_convT <<<(Hv * Hv + 255) / 256, 256, 0, stream>>>(Wk, WkT_h);
    k_bias2 <<<(G4H + 255) / 256, 256, 0, stream>>>(b_ih, b_hh, bsum, G4H);
    k_gather<<<(Tv * Bv * Dv + 255) / 256, 256, 0, stream>>>(x, teach, embed, emb_h);
    k_init  <<<(Bv * Hv + 255) / 256, 256, 0, stream>>>(cx, hx_h);

    // Gpre = emb @ W_ih^T + (b_ih + b_hh)   [4096 x 2048, K=512]
    gemm(emb_h, Dv, Wih_h, Dv, bsum, nullptr, 0, Gpre, G4H, Tv * Bv, G4H, Dv);
    // keys = z @ Wk                          [4096 x 512, K=512]
    gemm(z_h, Hv, WkT_h, Hv, nullptr, nullptr, 0, keys, Hv, Bv * Sv, Hv, Hv);

    // sequential decode
    for (int t = 0; t < Tv; ++t) {
        gemm(hx_h, Hv, Whh_h, Hv, nullptr, Gpre + (size_t)t * Bv * G4H, G4H,
             gates, G4H, Bv, G4H, Hv);
        k_cell<<<(Bv * Hv + 255) / 256, 256, 0, stream>>>(gates, cx, hx_h, hcat_h);
        gemm(hx_h, Hv, Wq_h, Hv, nullptr, nullptr, 0, qbuf, Hv, Bv, Hv, Hv);
        k_attn<<<Bv, Sv, 0, stream>>>(qbuf, keys, z, v_att, hcat_h);
        gemm(hcat_h, H2, Wp_h, H2, b_proj, nullptr, 0,
             out + (size_t)t * Vv, (long long)Tv * Vv, Bv, Vv, H2);
    }
}
